// LSTM_65395172049322
// MI455X (gfx1250) — matrix operand order repaired and checked by reading, not compiled, not run on hardware
//
#include <hip/hip_runtime.h>
#include <hip/hip_bf16.h>

// ---------------------------------------------------------------------------
// LSTM (T=512, B=256, D=256, H=256) for MI455X / gfx1250 (wave32, WMMA).
//
// Phase 1 (pack):   fp32 weights -> f16, pre-swizzled into WMMA B-fragment
//                   per-lane order; biases concatenated.
// Phase 2 (xgemm):  Xg[T*B,1024] = x @ [Wf|Wi|Wc|Wo](x-rows) + b, full-GPU
//                   parallel GEMM via v_wmma_f32_16x16x32_f16, stored f16 in
//                   C-fragment order (256 MB of d_ws).
// Phase 3 (recur):  16 persistent workgroups (one per 16-row batch tile),
//                   512 threads = 16 waves; wave w owns latent cols
//                   [16w,16w+16). Per step: acc = Xg + h @ Wh (32 WMMAs,
//                   Wh B-fragments resident in VGPRs), gates in fp32 VALU,
//                   h written to LDS in A-fragment order (double-buffered),
//                   one workgroup barrier per step. No global sync: the
//                   recurrence is independent per batch row.
//                   Xg is software-pipelined one step ahead into VGPRs and
//                   prefetched three steps ahead into L2 (global_prefetch_b8)
//                   so no step starts with an HBM-latency stall.
//
// d_ws requirement: 2 MB (packed weights/bias) + 256 MB (Xg) = ~258 MB.
// ---------------------------------------------------------------------------

typedef __attribute__((ext_vector_type(16))) _Float16 v16h;
typedef __attribute__((ext_vector_type(8)))  _Float16 v8h;
typedef __attribute__((ext_vector_type(8)))  float    v8f;
typedef __attribute__((ext_vector_type(4)))  float    v4f;

#define T_STEPS 512
#define BATCH   256
#define DDIM    256
#define DLAT    256
#define NCOLS   1024          // 4 gates * DLAT
#define NTILES  64            // NCOLS / 16
#define KT_REC  8             // 256 / 32 K-tiles for recurrent GEMM
#define MTILES  8192          // (T*B)/16

// d_ws byte offsets
#define WXP_OFF 0u                    // 64*8*32*16 halves = 512 KB (x-part weights)
#define WHP_OFF (512u*1024u)          // 512 KB (h-part weights)
#define BC_OFF  (1024u*1024u)         // 1024 floats = 4 KB (concat bias)
#define XG_OFF  (2u*1024u*1024u)      // 8192*64*32*8 halves = 256 MB

__device__ __forceinline__ float sigf(float x) {
    return 1.0f / (1.0f + __expf(-x));
}
__device__ __forceinline__ float tanhf_fast(float x) {
    float e = __expf(-2.0f * x);
    return (1.0f - e) / (1.0f + e);
}

// ---------------------------------------------------------------------------
// Kernel 1: pack weights (fp32 -> f16, WMMA B-fragment per-lane order) + bias.
// B fragment (nt, kt): lane l holds col n = nt*16 + (l&15),
// halves e=0..15 <-> K = kt*32 + (l>>4)*8 + (e&7) + (e>>3)*16.
// Packed idx: ((nt*8 + kt)*32 + lane)*16 + e.
// ---------------------------------------------------------------------------
__global__ __launch_bounds__(256) void lstm_pack_kernel(
    const float* __restrict__ Wf, const float* __restrict__ bf,
    const float* __restrict__ Wi, const float* __restrict__ bi,
    const float* __restrict__ Wc, const float* __restrict__ bc,
    const float* __restrict__ Wo, const float* __restrict__ bo,
    _Float16* __restrict__ wxp, _Float16* __restrict__ whp,
    float* __restrict__ bcat)
{
    unsigned idx = blockIdx.x * 256u + threadIdx.x;
    const unsigned NW = 2u * NTILES * KT_REC * 32u * 16u;   // 524288
    if (idx < NW) {
        unsigned e    = idx & 15u;
        unsigned lane = (idx >> 4) & 31u;
        unsigned kt   = (idx >> 9) & 7u;
        unsigned nt   = (idx >> 12) & 63u;
        unsigned which = idx >> 18;                 // 0 = x-part, 1 = h-part
        unsigned K   = kt * 32u + (lane >> 4) * 8u + (e & 7u) + (e >> 3) * 16u;      // 0..255
        unsigned g   = nt >> 4;
        unsigned lat = (nt & 15u) * 16u + (lane & 15u);       // 0..255
        unsigned ksrc = K + which * 256u;                     // row of W [512,256]
        const float* W = (g == 0) ? Wf : (g == 1) ? Wi : (g == 2) ? Wc : Wo;
        float v = W[ksrc * 256u + lat];
        _Float16* dst = which ? whp : wxp;
        dst[idx & 0x3FFFFu] = (_Float16)v;
    } else {
        unsigned t2 = idx - NW;                               // 0..1023
        unsigned g = t2 >> 8, l = t2 & 255u;
        const float* b = (g == 0) ? bf : (g == 1) ? bi : (g == 2) ? bc : bo;
        bcat[t2] = b[l];
    }
}

// ---------------------------------------------------------------------------
// Kernel 2: Xg = x @ Wx + bias, stored f16 in C-fragment order.
// One wave per (mtile, ngroup-of-4-ntiles): 131072 waves, 8 waves/block.
// A fragment built from fp32 x with v_cvt_pk_f16_f32.
// ---------------------------------------------------------------------------
__global__ __launch_bounds__(256) void lstm_xgemm_kernel(
    const float* __restrict__ x,
    const _Float16* __restrict__ wxp,
    const float* __restrict__ bcat,
    _Float16* __restrict__ xg)
{
    const unsigned lane = threadIdx.x & 31u;
    const unsigned gw   = blockIdx.x * 8u + (threadIdx.x >> 5);
    const unsigned mtile = gw >> 4;            // 0..8191
    const unsigned ng    = gw & 15u;           // ntiles ng*4 .. ng*4+3

    const unsigned row   = mtile * 16u + (lane & 15u);       // row of x [T*B, 256]
    const unsigned koff  = (lane >> 4) * 8u;                 // 0 or 8

    v8f acc0 = {}, acc1 = {}, acc2 = {}, acc3 = {};

    #pragma unroll
    for (int kt = 0; kt < KT_REC; ++kt) {
        // A fragment: halves 0..7 = K kt*32+koff.. , halves 8..15 = +16
        const float* rp = x + (size_t)row * DDIM + kt * 32u + koff;
        v4f f0 = *(const v4f*)(rp + 0);
        v4f f1 = *(const v4f*)(rp + 4);
        v4f f2 = *(const v4f*)(rp + 16);
        v4f f3 = *(const v4f*)(rp + 20);
        v16h a;
        #pragma unroll
        for (int i = 0; i < 4; ++i) {
            a[i]      = (_Float16)f0[i];
            a[4 + i]  = (_Float16)f1[i];
            a[8 + i]  = (_Float16)f2[i];
            a[12 + i] = (_Float16)f3[i];
        }
        #pragma unroll
        for (int j = 0; j < 4; ++j) {
            unsigned nt = ng * 4u + j;
            v16h b = *(const v16h*)&wxp[(((size_t)nt * KT_REC + kt) * 32u + lane) * 16u];
            if (j == 0) acc0 = __builtin_amdgcn_wmma_f32_16x16x32_f16(false, a, false, b, (short)0, acc0, false, false);
            if (j == 1) acc1 = __builtin_amdgcn_wmma_f32_16x16x32_f16(false, a, false, b, (short)0, acc1, false, false);
            if (j == 2) acc2 = __builtin_amdgcn_wmma_f32_16x16x32_f16(false, a, false, b, (short)0, acc2, false, false);
            if (j == 3) acc3 = __builtin_amdgcn_wmma_f32_16x16x32_f16(false, a, false, b, (short)0, acc3, false, false);
        }
    }

    // bias + convert + store (f16, fragment order: ((mtile*64+nt)*32+lane)*8)
    #pragma unroll
    for (int j = 0; j < 4; ++j) {
        unsigned nt = ng * 4u + j;
        float bv = bcat[(nt >> 4) * 256u + (nt & 15u) * 16u + (lane & 15u)];
        v8f acc = (j == 0) ? acc0 : (j == 1) ? acc1 : (j == 2) ? acc2 : acc3;
        v8h h;
        #pragma unroll
        for (int v = 0; v < 8; ++v) h[v] = (_Float16)(acc[v] + bv);
        *(v8h*)&xg[(((size_t)mtile * NTILES + nt) * 32u + lane) * 8u] = h;
    }
}

// ---------------------------------------------------------------------------
// Kernel 3: persistent recurrence. 16 blocks (batch tiles) x 512 threads
// (16 waves). Wave w owns latent tile w; its 4 gate-tile B-fragments of Wh
// (256 VGPRs) stay resident across all 512 steps. h is double-buffered in
// LDS in A-fragment order. Xg is register-pipelined one step ahead and
// L2-prefetched three steps ahead so the WMMA chain never waits on HBM.
// ---------------------------------------------------------------------------
__global__ __launch_bounds__(512, 1) void lstm_recurrent_kernel(
    const _Float16* __restrict__ whp,
    const _Float16* __restrict__ xg,
    float* __restrict__ out)
{
    const unsigned lane = threadIdx.x & 31u;
    const unsigned lt   = threadIdx.x >> 5;     // latent tile 0..15
    const unsigned bt   = blockIdx.x;           // batch tile 0..15

    // Resident Wh B-fragments: wf[g][kt]
    v16h wf[4][KT_REC];
    #pragma unroll
    for (int g = 0; g < 4; ++g) {
        unsigned nt = g * 16u + lt;
        #pragma unroll
        for (int kt = 0; kt < KT_REC; ++kt)
            wf[g][kt] = *(const v16h*)&whp[(((size_t)nt * KT_REC + kt) * 32u + lane) * 16u];
    }

    // Per-wave base into Xg for gate g at mtile: xgw + (mtile*64 + g*16)*256
    // (each fragment slot is 32 lanes * 8 halves; lane offset folded in)
    const _Float16* xgw = xg + ((size_t)lt * 32u + lane) * 8u;
    #define XG_AT(mt, g) (xgw + (((size_t)(mt) * NTILES + (g) * 16u) * 32u) * 8u)

    // h double buffer, A-fragment order: hb[buf][(kt*32+lane)*16 + e]
    __shared__ _Float16 hb[2][KT_REC * 32 * 16];
    #pragma unroll
    for (int i = 0; i < 8; ++i)
        hb[0][threadIdx.x + i * 512u] = (_Float16)0.0f;
    __syncthreads();

    float cst[8];
    #pragma unroll
    for (int v = 0; v < 8; ++v) cst[v] = 0.0f;

    const unsigned m = (lane >> 4) * 8u;                    // C layout: M = v + 8*(lane>=16)
    const unsigned Kcol = lt * 16u + (lane & 15u);          // this lane's latent column as K index
    const unsigned kt2 = Kcol >> 5;
    const unsigned r   = Kcol & 31u;
    const unsigned lanep_base = ((r >> 3) & 1u) ? 16u : 0u; // +16 for sub-chunks 1,3
    const unsigned e2  = (r & 7u) + ((r >> 4) ? 8u : 0u);   // half slot

    // Prologue: fetch Xg for t=0 into registers
    v8h xv[4];
    #pragma unroll
    for (int g = 0; g < 4; ++g) xv[g] = *(const v8h*)XG_AT(bt, g);

    for (int t = 0; t < T_STEPS; ++t) {
        const unsigned cur = t & 1, nxt = cur ^ 1;

        // Issue next step's Xg loads now (consumed next iteration -> latency
        // hidden behind this step's WMMA chain + elementwise work).
        const unsigned tn = (t < T_STEPS - 1) ? (unsigned)(t + 1) : (unsigned)t;
        const unsigned mtile_n = tn * 16u + bt;
        v8h xn[4];
        #pragma unroll
        for (int g = 0; g < 4; ++g) xn[g] = *(const v8h*)XG_AT(mtile_n, g);

        // L2 prefetch three steps ahead (global_prefetch_b8).
        const unsigned tp = (t + 3 < T_STEPS) ? (unsigned)(t + 3) : (unsigned)(T_STEPS - 1);
        const unsigned mtile_p = tp * 16u + bt;
        #pragma unroll
        for (int g = 0; g < 4; ++g) __builtin_prefetch(XG_AT(mtile_p, g), 0, 0);

        // accumulators start from this step's Xg (x-part + bias)
        v8f acc[4];
        #pragma unroll
        for (int g = 0; g < 4; ++g) {
            #pragma unroll
            for (int v = 0; v < 8; ++v) acc[g][v] = (float)xv[g][v];
        }

        // recurrent GEMM: + h @ Wh
        #pragma unroll
        for (int kt = 0; kt < KT_REC; ++kt) {
            v16h a = *(const v16h*)&hb[cur][((unsigned)kt * 32u + lane) * 16u];
            #pragma unroll
            for (int g = 0; g < 4; ++g)
                acc[g] = __builtin_amdgcn_wmma_f32_16x16x32_f16(
                    false, a, false, wf[g][kt], (short)0, acc[g], false, false);
        }

        // gates + state update + output, all per-lane fp32
        #pragma unroll
        for (int v = 0; v < 8; ++v) {
            float fg = sigf(acc[0][v]);
            float ig = sigf(acc[1][v]);
            float gg = tanhf_fast(acc[2][v]);
            float og = sigf(acc[3][v]);
            float c  = fg * cst[v] + ig * gg;
            cst[v] = c;
            float h = og * tanhf_fast(c);

            unsigned mrow = (unsigned)v + m;                               // 0..15
            // out[t][bt*16+mrow][lt*16 + lane%16]
            out[(((size_t)t * BATCH) + bt * 16u + mrow) * DLAT + lt * 16u + (lane & 15u)] = h;
            // h -> LDS next buffer, A-fragment order
            hb[nxt][((kt2 * 32u) + mrow + lanep_base) * 16u + e2] = (_Float16)h;
        }

        // rotate the Xg pipeline
        #pragma unroll
        for (int g = 0; g < 4; ++g) xv[g] = xn[g];

        __syncthreads();
    }
    #undef XG_AT
}

// ---------------------------------------------------------------------------
extern "C" void kernel_launch(void* const* d_in, const int* in_sizes, int n_in,
                              void* d_out, int out_size, void* d_ws, size_t ws_size,
                              hipStream_t stream) {
    const float* x  = (const float*)d_in[0];
    const float* Wf = (const float*)d_in[1];
    const float* bf = (const float*)d_in[2];
    const float* Wi = (const float*)d_in[3];
    const float* bi = (const float*)d_in[4];
    const float* Wc = (const float*)d_in[5];
    const float* bc = (const float*)d_in[6];
    const float* Wo = (const float*)d_in[7];
    const float* bo = (const float*)d_in[8];

    char* ws = (char*)d_ws;
    _Float16* wxp  = (_Float16*)(ws + WXP_OFF);
    _Float16* whp  = (_Float16*)(ws + WHP_OFF);
    float*    bcat = (float*)   (ws + BC_OFF);
    _Float16* xg   = (_Float16*)(ws + XG_OFF);

    // Phase 1: pack weights/bias (524288 + 1024 threads)
    lstm_pack_kernel<<<2052, 256, 0, stream>>>(Wf, bf, Wi, bi, Wc, bc, Wo, bo,
                                               wxp, whp, bcat);
    // Phase 2: parallel x-GEMM, 131072 waves = 16384 blocks x 256
    lstm_xgemm_kernel<<<16384, 256, 0, stream>>>(x, wxp, bcat, xg);
    // Phase 3: batch-parallel recurrence, 16 persistent workgroups
    lstm_recurrent_kernel<<<16, 512, 0, stream>>>(whp, xg, (float*)d_out);
}
